// RGAT_50362786513038
// MI455X (gfx1250) — matrix operand order repaired and checked by reading, not compiled, not run on hardware
//
#include <hip/hip_runtime.h>
#include <hip/hip_bf16.h>
#include <stdint.h>

// ---------------------------------------------------------------------------
// RGAT (2-layer relational GATv2), N=50000, H=4, R=3, E=524288, F=128
// Dense GEMMs on the CDNA5 bf16 WMMA path (weights pre-transposed so both
// A and B fragments are contiguous b128 loads); edge/softmax phase is
// bandwidth-bound streaming + f32 atomics.
// ---------------------------------------------------------------------------

#define HEADS 4

typedef __attribute__((ext_vector_type(16))) __bf16 v16bf;
typedef __attribute__((ext_vector_type(8)))  float  v8f;

// ------------------------- small utility kernels ---------------------------

__global__ void fill_f32_kernel(float* __restrict__ p, float v, size_t n) {
    size_t i = (size_t)blockIdx.x * blockDim.x + threadIdx.x;
    if (i < n) p[i] = v;
}

__global__ void fill_u32_kernel(unsigned int* __restrict__ p, unsigned int v, size_t n) {
    size_t i = (size_t)blockIdx.x * blockDim.x + threadIdx.x;
    if (i < n) p[i] = v;
}

__device__ __forceinline__ unsigned short f32_to_bf16_rne(float f) {
    unsigned int u = __float_as_uint(f);
    u += 0x7FFFu + ((u >> 16) & 1u);   // round-to-nearest-even
    return (unsigned short)(u >> 16);
}

__global__ void cvt_f32_bf16_kernel(const float* __restrict__ in,
                                    unsigned short* __restrict__ out, size_t n) {
    size_t i = (size_t)blockIdx.x * blockDim.x + threadIdx.x;
    if (i < n) out[i] = f32_to_bf16_rne(in[i]);
}

// Convert weights [R, K, Nc] f32 -> transposed bf16 [R, Nc, K] so that the
// WMMA B fragment (contiguous K run per column) loads as global_load_b128.
__global__ void cvt_w_bf16_T_kernel(const float* __restrict__ in,
                                    unsigned short* __restrict__ out,
                                    int K, int Nc, size_t total) {
    size_t i = (size_t)blockIdx.x * blockDim.x + threadIdx.x;
    if (i >= total) return;
    const size_t mat = (size_t)K * Nc;
    const size_t r   = i / mat;
    const size_t j   = i - r * mat;
    const int    n   = (int)(j / K);
    const int    k   = (int)(j - (size_t)n * K);
    out[i] = f32_to_bf16_rne(in[r * mat + (size_t)k * Nc + n]);
}

__global__ void relu_f32_to_bf16_kernel(const float* __restrict__ in,
                                        unsigned short* __restrict__ out, size_t n) {
    size_t i = (size_t)blockIdx.x * blockDim.x + threadIdx.x;
    if (i < n) {
        float v = in[i];
        out[i] = f32_to_bf16_rne(v > 0.f ? v : 0.f);
    }
}

// ----------------------------- WMMA GEMM -----------------------------------
// C[M,Nc] (f32) = A[M,K] (bf16 row-major) * B (bf16, TRANSPOSED: [Nc,K])
// One wave computes a 16x64 output tile: 4 accumulators sharing one A
// fragment per 32-wide K step -> 4 independent v_wmma_f32_16x16x32_bf16
// back-to-back (hides the bf16 WMMA RAW hazard window).
//
// Fragment layouts (ISA 7.12.2, wave32):
//   A 16x32: lane l(0..15) row M=l; half-wave h=0: K {0..7,16..23},
//            h=1: K {8..15,24..31} -> two contiguous 16B chunks.
//   B 32x16: lane l column N=l; half-wave h: K {8h..8h+7, 16+8h..16+8h+7}
//            -> two contiguous 16B chunks in the transposed weight.
//   C/D:     VGPR r -> row M = r + 8*h, col N = lane&15.
__global__ void gemm_bf16_wmma_kernel(const unsigned short* __restrict__ A,
                                      const unsigned short* __restrict__ BT,
                                      float* __restrict__ C,
                                      int M, int K, int Nc) {
    const int wave  = threadIdx.x >> 5;
    const int lane  = threadIdx.x & 31;
    const int tileM = (blockIdx.x * 4 + wave) * 16;     // wave-uniform
    if (tileM >= M) return;                             // uniform branch
    const int tileN = blockIdx.y * 64;

    const int h = lane >> 4;          // half-wave: 0 or 1
    const int l = lane & 15;
    const long rowA = (long)tileM + l;

    union Frag { uint4 q[2]; v16bf v; };

    v8f acc[4] = {{}, {}, {}, {}};

    const unsigned short* pa = A + rowA * K;

    for (int k0 = 0; k0 < K; k0 += 32) {
        Frag fa;
        fa.q[0] = *(const uint4*)(pa + k0 + h * 8);
        fa.q[1] = *(const uint4*)(pa + k0 + 16 + h * 8);

        Frag fb[4];
#pragma unroll
        for (int c = 0; c < 4; ++c) {
            const unsigned short* pb =
                BT + (size_t)(tileN + 16 * c + l) * K + k0 + h * 8;
            fb[c].q[0] = *(const uint4*)(pb);
            fb[c].q[1] = *(const uint4*)(pb + 16);
        }
#pragma unroll
        for (int c = 0; c < 4; ++c) {
            acc[c] = __builtin_amdgcn_wmma_f32_16x16x32_bf16(
                         false, fa.v, false, fb[c].v, (short)0, acc[c],
                         false, false);
        }
    }

#pragma unroll
    for (int c = 0; c < 4; ++c) {
        const int col = tileN + 16 * c + l;
#pragma unroll
        for (int r = 0; r < 8; ++r) {
            const long orow = (long)tileM + r + 8 * h;
            C[orow * Nc + col] = acc[c][r];
        }
    }
}

// ------------------------- edge / softmax kernels --------------------------

__device__ __forceinline__ unsigned int flip_f32(float x) {
    unsigned int u = __float_as_uint(x);
    return (u & 0x80000000u) ? ~u : (u | 0x80000000u);
}
__device__ __forceinline__ float unflip_f32(unsigned int u) {
    u = (u & 0x80000000u) ? (u & 0x7FFFFFFFu) : ~u;
    return __uint_as_float(u);
}

// e[e,h] = sum_d leaky_relu(hs[src,h,d] + hd[dst,h,d], 0.2) * a[h,d]
// and segment-max via ordered-uint atomicMax (emax_u initialized to 0,
// which is below flip() of every finite float).
__global__ void edge_score_max_kernel(const float* __restrict__ hs,
                                      const float* __restrict__ hd,
                                      const float* __restrict__ avec,
                                      const int* __restrict__ src,
                                      const int* __restrict__ dst,
                                      float* __restrict__ escore,
                                      unsigned int* __restrict__ emax_u,
                                      int E, int D) {
    int idx = blockIdx.x * blockDim.x + threadIdx.x;
    if (idx >= E * HEADS) return;
    const int e  = idx >> 2;        // HEADS == 4
    const int hh = idx & 3;
    const int s = src[e];
    const int d = dst[e];
    const float4* ps = (const float4*)(hs + (size_t)s * (HEADS * D) + hh * D);
    const float4* pd = (const float4*)(hd + (size_t)d * (HEADS * D) + hh * D);
    const float4* pa = (const float4*)(avec + hh * D);
    float acc = 0.f;
    const int q = D >> 2;
    for (int j = 0; j < q; ++j) {
        float4 vs = ps[j], vd = pd[j], va = pa[j];
        float m0 = vs.x + vd.x, m1 = vs.y + vd.y;
        float m2 = vs.z + vd.z, m3 = vs.w + vd.w;
        m0 = (m0 > 0.f) ? m0 : 0.2f * m0;
        m1 = (m1 > 0.f) ? m1 : 0.2f * m1;
        m2 = (m2 > 0.f) ? m2 : 0.2f * m2;
        m3 = (m3 > 0.f) ? m3 : 0.2f * m3;
        acc += m0 * va.x + m1 * va.y + m2 * va.z + m3 * va.w;
    }
    escore[idx] = acc;
    atomicMax(&emax_u[(size_t)d * HEADS + hh], flip_f32(acc));
}

// ee = exp(e - emax[dst]); den[dst] += ee  (escore overwritten with ee)
__global__ void edge_exp_den_kernel(float* __restrict__ escore,
                                    const int* __restrict__ dst,
                                    const unsigned int* __restrict__ emax_u,
                                    float* __restrict__ den,
                                    int E) {
    int idx = blockIdx.x * blockDim.x + threadIdx.x;
    if (idx >= E * HEADS) return;
    const int e  = idx >> 2;
    const int hh = idx & 3;
    const int d = dst[e];
    const float mx = unflip_f32(emax_u[(size_t)d * HEADS + hh]);
    const float ee = __expf(escore[idx] - mx);
    escore[idx] = ee;
    atomicAdd(&den[(size_t)d * HEADS + hh], ee);
}

// out[dst,h,:] += (ee/(den[dst,h]+1e-9)) * hs[src,h,:]
__global__ void edge_aggregate_kernel(const float* __restrict__ hs,
                                      const float* __restrict__ escore,
                                      const float* __restrict__ den,
                                      const int* __restrict__ src,
                                      const int* __restrict__ dst,
                                      float* __restrict__ out,
                                      int E, int D) {
    int idx = blockIdx.x * blockDim.x + threadIdx.x;
    if (idx >= E * HEADS) return;
    const int e  = idx >> 2;
    const int hh = idx & 3;
    const int s = src[e];
    const int d = dst[e];
    const float alpha = escore[idx] / (den[(size_t)d * HEADS + hh] + 1e-9f);
    const float4* ps = (const float4*)(hs + (size_t)s * (HEADS * D) + hh * D);
    float* po = out + (size_t)d * (HEADS * D) + hh * D;
    const int q = D >> 2;
    for (int j = 0; j < q; ++j) {
        float4 v = ps[j];
        atomicAdd(&po[4 * j + 0], alpha * v.x);
        atomicAdd(&po[4 * j + 1], alpha * v.y);
        atomicAdd(&po[4 * j + 2], alpha * v.z);
        atomicAdd(&po[4 * j + 3], alpha * v.w);
    }
}

// ------------------------------- driver ------------------------------------

static inline size_t ws_align(size_t x) { return (x + 255) & ~(size_t)255; }

extern "C" void kernel_launch(void* const* d_in, const int* in_sizes, int n_in,
                              void* d_out, int out_size, void* d_ws, size_t ws_size,
                              hipStream_t stream) {
    (void)n_in; (void)out_size; (void)ws_size;

    const float* x   = (const float*)d_in[0];
    const int*   src = (const int*)d_in[1];
    const int*   dst = (const int*)d_in[2];
    const float* W1s = (const float*)d_in[3];
    const float* W1d = (const float*)d_in[4];
    const float* a1  = (const float*)d_in[5];
    const float* W2s = (const float*)d_in[6];
    const float* W2d = (const float*)d_in[7];
    const float* a2  = (const float*)d_in[8];

    const int Nn = in_sizes[0] / 128;   // 50000 nodes
    const int R  = 3;
    const int E  = in_sizes[1] / R;     // 524288 edges per relation
    const int F  = 128;                 // input feature dim
    const int H1 = 128;                 // layer-1 width (H*32)
    const int H2 = 256;                 // layer-2 width (H*64)

    // ------------------ carve workspace ------------------
    char* p = (char*)d_ws;
    auto carve = [&](size_t bytes) { char* q = p; p += ws_align(bytes); return q; };

    unsigned short* xbf   = (unsigned short*)carve((size_t)Nn * F * 2);
    unsigned short* w1sbf = (unsigned short*)carve((size_t)R * F * H1 * 2);  // [R,H1,F] transposed
    unsigned short* w1dbf = (unsigned short*)carve((size_t)R * F * H1 * 2);
    unsigned short* w2sbf = (unsigned short*)carve((size_t)R * H1 * H2 * 2); // [R,H2,H1] transposed
    unsigned short* w2dbf = (unsigned short*)carve((size_t)R * H1 * H2 * 2);
    float*          hsbuf = (float*)carve((size_t)Nn * H2 * 4);   // max width
    float*          hdbuf = (float*)carve((size_t)Nn * H2 * 4);
    float*          escr  = (float*)carve((size_t)E * HEADS * 4);
    unsigned int*   emaxu = (unsigned int*)carve((size_t)Nn * HEADS * 4);
    float*          den   = (float*)carve((size_t)Nn * HEADS * 4);
    float*          h1acc = (float*)carve((size_t)Nn * H1 * 4);
    unsigned short* h1bf  = (unsigned short*)carve((size_t)Nn * H1 * 2);

    const int TB = 256;
    auto nb = [](size_t n, int tb) { return (int)((n + tb - 1) / tb); };

    // ------------------ precision conversion ------------------
    cvt_f32_bf16_kernel<<<nb((size_t)Nn * F, TB), TB, 0, stream>>>(x, xbf, (size_t)Nn * F);
    {
        const size_t t1 = (size_t)R * F * H1;
        cvt_w_bf16_T_kernel<<<nb(t1, TB), TB, 0, stream>>>(W1s, w1sbf, F, H1, t1);
        cvt_w_bf16_T_kernel<<<nb(t1, TB), TB, 0, stream>>>(W1d, w1dbf, F, H1, t1);
        const size_t t2 = (size_t)R * H1 * H2;
        cvt_w_bf16_T_kernel<<<nb(t2, TB), TB, 0, stream>>>(W2s, w2sbf, H1, H2, t2);
        cvt_w_bf16_T_kernel<<<nb(t2, TB), TB, 0, stream>>>(W2d, w2dbf, H1, H2, t2);
    }

    const int mTiles = (Nn + 15) / 16;           // 3125
    const int gemmGX = (mTiles + 3) / 4;         // 4 waves / block
    const int eThreads = E * HEADS;

    // ====================== LAYER 1 ======================
    fill_f32_kernel<<<nb((size_t)Nn * H1, TB), TB, 0, stream>>>(h1acc, 0.f, (size_t)Nn * H1);

    for (int r = 0; r < R; ++r) {
        const int* src_r = src + (size_t)r * E;
        const int* dst_r = dst + (size_t)r * E;

        dim3 gg(gemmGX, H1 / 64);
        gemm_bf16_wmma_kernel<<<gg, 128, 0, stream>>>(
            xbf, w1sbf + (size_t)r * F * H1, hsbuf, Nn, F, H1);
        gemm_bf16_wmma_kernel<<<gg, 128, 0, stream>>>(
            xbf, w1dbf + (size_t)r * F * H1, hdbuf, Nn, F, H1);

        fill_u32_kernel<<<nb((size_t)Nn * HEADS, TB), TB, 0, stream>>>(emaxu, 0u, (size_t)Nn * HEADS);
        fill_f32_kernel<<<nb((size_t)Nn * HEADS, TB), TB, 0, stream>>>(den, 0.f, (size_t)Nn * HEADS);

        edge_score_max_kernel<<<nb(eThreads, TB), TB, 0, stream>>>(
            hsbuf, hdbuf, a1 + (size_t)r * HEADS * 32, src_r, dst_r,
            escr, emaxu, E, 32);
        edge_exp_den_kernel<<<nb(eThreads, TB), TB, 0, stream>>>(
            escr, dst_r, emaxu, den, E);
        edge_aggregate_kernel<<<nb(eThreads, TB), TB, 0, stream>>>(
            hsbuf, escr, den, src_r, dst_r, h1acc, E, 32);
    }

    // relu + cast to bf16 for layer-2 GEMMs
    relu_f32_to_bf16_kernel<<<nb((size_t)Nn * H1, TB), TB, 0, stream>>>(
        h1acc, h1bf, (size_t)Nn * H1);

    // ====================== LAYER 2 ======================
    float* outf = (float*)d_out;   // [Nn, H2] f32
    fill_f32_kernel<<<nb((size_t)Nn * H2, TB), TB, 0, stream>>>(outf, 0.f, (size_t)Nn * H2);

    for (int r = 0; r < R; ++r) {
        const int* src_r = src + (size_t)r * E;
        const int* dst_r = dst + (size_t)r * E;

        dim3 gg(gemmGX, H2 / 64);
        gemm_bf16_wmma_kernel<<<gg, 128, 0, stream>>>(
            h1bf, w2sbf + (size_t)r * H1 * H2, hsbuf, Nn, H1, H2);
        gemm_bf16_wmma_kernel<<<gg, 128, 0, stream>>>(
            h1bf, w2dbf + (size_t)r * H1 * H2, hdbuf, Nn, H1, H2);

        fill_u32_kernel<<<nb((size_t)Nn * HEADS, TB), TB, 0, stream>>>(emaxu, 0u, (size_t)Nn * HEADS);
        fill_f32_kernel<<<nb((size_t)Nn * HEADS, TB), TB, 0, stream>>>(den, 0.f, (size_t)Nn * HEADS);

        edge_score_max_kernel<<<nb(eThreads, TB), TB, 0, stream>>>(
            hsbuf, hdbuf, a2 + (size_t)r * HEADS * 64, src_r, dst_r,
            escr, emaxu, E, 64);
        edge_exp_den_kernel<<<nb(eThreads, TB), TB, 0, stream>>>(
            escr, dst_r, emaxu, den, E);
        edge_aggregate_kernel<<<nb(eThreads, TB), TB, 0, stream>>>(
            hsbuf, escr, den, src_r, dst_r, outf, E, 64);
    }
}
